// GraphAttnBias_28930899706439
// MI455X (gfx1250) — matrix operand order repaired and checked by reading, not compiled, not run on hardware
//
#include <hip/hip_runtime.h>
#include <hip/hip_bf16.h>

typedef __attribute__((ext_vector_type(16))) _Float16 v16h;
typedef __attribute__((ext_vector_type(8)))  _Float16 v8h;
typedef __attribute__((ext_vector_type(8)))  float    v8f;

#define NROWS   (16*128)     // B*N
#define NP1     129
#define PLANE   (129*129)

// Main kernel: computes out[b, h, 1+i, 1+j] = 2*attn_bias + sp_bias + edge_b
// One wave handles a 16-wide j tile; 8 waves/block = one full (b,i) row per pass.
__global__ __launch_bounds__(256) void graph_bias_main(
    const float* __restrict__ attn_bias,     // (B,N,N) f32
    const int*   __restrict__ spatial_pos,   // (B,N,N) i32
    const int*   __restrict__ edge_input,    // (B,N,N,5,3) i32
    const float* __restrict__ edge_encoder_w,// (513,32) f32
    const float* __restrict__ spatial_pos_w, // (512,32) f32
    const float* __restrict__ edge_dis_w,    // (>=5,32,32) f32
    float*       __restrict__ out)           // (B,32,129,129) f32
{
    __shared__ __attribute__((aligned(16))) _Float16 ee[513*32];   // f16 edge-encoder table
    __shared__ __attribute__((aligned(16))) float    tile[8][32*21]; // per-wave transpose buf

    const int tid = threadIdx.x;
    // Stage edge_encoder_w into LDS as f16
    for (int t = tid; t < 513*32; t += 256) ee[t] = (_Float16)edge_encoder_w[t];
    __syncthreads();

    const int lane = tid & 31;
    const int wave = tid >> 5;
    const int col  = lane & 15;   // A row / D column index owned by this lane
    const int hi   = lane >> 4;   // half-wave selector

    // ---- Load B matrices (dis_w[d], split into two 16-col halves) into registers.
    // B (32x16 f16): lane = column n (l&15); halfs t<8 -> K = 8*(l>>4) + t, t>=8 -> K = 16 + 8*(l>>4) + (t-8).
    v16h Bv[5][2];
    #pragma unroll
    for (int d = 0; d < 5; ++d)
        #pragma unroll
        for (int nh = 0; nh < 2; ++nh)
            #pragma unroll
            for (int t = 0; t < 16; ++t) {
                int h = (t < 8) ? (hi*8 + t) : (16 + hi*8 + (t - 8));            // K index (head-in)
                int n = col + nh*16;          // output head index
                Bv[d][nh][t] = (_Float16)edge_dis_w[d*1024 + h*32 + n];
            }

    float* tl = tile[wave];
    const int j0 = wave * 16;

    for (int bi = blockIdx.x; bi < NROWS; bi += gridDim.x) {
        const int bb = bi >> 7;
        const int ii = bi & 127;
        const size_t rowbase = (size_t)bi * 128;   // (b*128 + i) * 128

        // ---- Per-lane pair p = col: its 15 edge indices + sp divisor
        const int* eip = edge_input + (rowbase + (size_t)(j0 + col)) * 15;
        int eidx[15];
        #pragma unroll
        for (int k = 0; k < 15; ++k) eidx[k] = eip[k];

        int spv = spatial_pos[rowbase + j0 + col];
        int sp  = (spv == 0) ? 1 : ((spv > 1) ? spv - 1 : spv);
        sp = (sp > 5) ? 5 : sp;
        _Float16 hr = (_Float16)(1.0f / (float)sp);
        v8h hs;
        #pragma unroll
        for (int t = 0; t < 8; ++t) hs[t] = hr;

        // ---- C init: c[m][n] = 2*attn_bias[b,i,j0+m] + spatial_pos_w[sp_idx][n]
        v8f c0, c1;
        const float* abr = attn_bias   + rowbase + j0;
        const int*   spr = spatial_pos + rowbase + j0;
        #pragma unroll
        for (int r = 0; r < 8; ++r) {
            int m = r + hi*8;
            float abv = 2.0f * abr[m];
            const float* spw = spatial_pos_w + (size_t)spr[m] * 32;
            c0[r] = abv + spw[col];
            c1[r] = abv + spw[col + 16];
        }

        // ---- 5 accumulating WMMAs per n-half: D += (emb_d / sp) x dis_w[d]
        // A (16x32 f16): lane row = col; halfs 0..7 -> K = hi*8 + t,
        //                halfs 8..15 -> K = 16 + hi*8 + (t-8).
        #pragma unroll
        for (int d = 0; d < 5; ++d) {
            const v8h* p0 = (const v8h*)(ee + (size_t)eidx[d*3+0]*32 + hi*8);
            const v8h* p1 = (const v8h*)(ee + (size_t)eidx[d*3+1]*32 + hi*8);
            const v8h* p2 = (const v8h*)(ee + (size_t)eidx[d*3+2]*32 + hi*8);
            v8h r0 = (p0[0] + p1[0] + p2[0]) * hs;  // K = hi*8 .. hi*8+7
            v8h r1 = (p0[2] + p1[2] + p2[2]) * hs;  // K = 16+hi*8 .. +7 (offset 16 halfs)
            v16h a;
            #pragma unroll
            for (int t = 0; t < 8; ++t) { a[t] = r0[t]; a[t+8] = r1[t]; }
            c0 = __builtin_amdgcn_wmma_f32_16x16x32_f16(false, a, false, Bv[d][0],
                                                        (short)0, c0, false, false);
            c1 = __builtin_amdgcn_wmma_f32_16x16x32_f16(false, a, false, Bv[d][1],
                                                        (short)0, c1, false, false);
        }

        // ---- Transpose D (m-major per head) through LDS, then j-contiguous stores.
        // Layout tl[h*21 + m]; stride 21 is conflict-free for both phases.
        #pragma unroll
        for (int r = 0; r < 8; ++r) {
            int m = r + hi*8;
            tl[col*21 + m]        = c0[r];
            tl[(col + 16)*21 + m] = c1[r];
        }
        asm volatile("s_wait_dscnt 0x0" ::: "memory");

        const size_t ob = (size_t)bb * 32 * PLANE + (size_t)(ii + 1) * NP1 + (j0 + 1);
        #pragma unroll
        for (int q = 0; q < 4; ++q) {
            int h  = q*8 + (lane >> 2);   // 8 heads per pass, 4 lanes each
            int mb = lane & 3;
            float* op = out + ob + (size_t)h * PLANE;
            #pragma unroll
            for (int k = 0; k < 4; ++k)   // lanes 0..3 contiguous in j
                op[mb + k*4] = tl[h*21 + mb + k*4];
        }
        // LDS ops are in-order per wave -> next iteration's tl writes are WAR-safe.
    }
}

// Border kernel: row 0 and column 0 get virt_w[h]; corner (0,0) = 0.
__global__ __launch_bounds__(256) void graph_bias_border(
    float* __restrict__ out, const float* __restrict__ virt_w)
{
    int tid = blockIdx.x * blockDim.x + threadIdx.x;
    const int total = 16 * 32 * NP1;
    if (tid >= total) return;
    int c  = tid % NP1;
    int bh = tid / NP1;
    int h  = bh & 31;
    float t = virt_w[h];
    size_t base = (size_t)bh * PLANE;
    out[base + c] = (c == 0) ? 0.0f : t;          // row 0
    if (c >= 1) out[base + (size_t)c * NP1] = t;  // column 0 (rows 1..128)
}

extern "C" void kernel_launch(void* const* d_in, const int* in_sizes, int n_in,
                              void* d_out, int out_size, void* d_ws, size_t ws_size,
                              hipStream_t stream) {
    const float* attn_bias      = (const float*)d_in[0];
    const int*   spatial_pos    = (const int*)  d_in[1];
    // d_in[2] = x (unused), d_in[4] = attn_edge_type (unused by reference)
    const int*   edge_input     = (const int*)  d_in[3];
    const float* edge_encoder_w = (const float*)d_in[5];
    const float* spatial_pos_w  = (const float*)d_in[6];
    const float* edge_dis_w     = (const float*)d_in[7];
    const float* virt_w         = (const float*)d_in[8];
    float* out = (float*)d_out;

    graph_bias_main<<<dim3(256), dim3(256), 0, stream>>>(
        attn_bias, spatial_pos, edge_input,
        edge_encoder_w, spatial_pos_w, edge_dis_w, out);

    int nthr = 16 * 32 * NP1;
    graph_bias_border<<<dim3((nthr + 255) / 256), dim3(256), 0, stream>>>(out, virt_w);
}
